// GNNTorchModel_49349174231512
// MI455X (gfx1250) — matrix operand order repaired and checked by reading, not compiled, not run on hardware
//
#include <hip/hip_runtime.h>
#include <math.h>

// ---------------------------------------------------------------- constants
#define NN 50000      // nodes
#define NE 800000     // edges
#define FDIM 32       // raw(16) + emb(16), bf16 feature stride
#define HDIM 64       // hidden per node

typedef __attribute__((ext_vector_type(16))) __bf16 v16bf;
typedef __attribute__((ext_vector_type(8)))  __bf16 v8bf;
typedef __attribute__((ext_vector_type(8)))  float  v8f;

// --------------------------------------------------------- WMMA fragment IO
// 16-bit A matrix 16x32 (MxK), row-major LDS source.
__device__ __forceinline__ v16bf load_a_frag(const __bf16* X, int stride,
                                             int kbase, int lane) {
  const int m = lane & 15, half = lane >> 4;
  const __bf16* row = X + m * stride + kbase;
  v8bf lo = *reinterpret_cast<const v8bf*>(row + 8 * half);
  v8bf hi = *reinterpret_cast<const v8bf*>(row + 16 + 8 * half);
  return __builtin_shufflevector(lo, hi, 0,1,2,3,4,5,6,7,8,9,10,11,12,13,14,15);
}

// 16-bit B matrix 32x16 (KxN) from W^T stored row-major [N x Kpad].
__device__ __forceinline__ v16bf load_b_frag(const __bf16* WtTile, int stride,
                                             int kbase, int lane) {
  const int n = lane & 15, half = lane >> 4;
  const __bf16* p = WtTile + n * stride + kbase;
  v8bf lo = *reinterpret_cast<const v8bf*>(p + 8 * half);
  v8bf hi = *reinterpret_cast<const v8bf*>(p + 16 + 8 * half);
  return __builtin_shufflevector(lo, hi, 0,1,2,3,4,5,6,7,8,9,10,11,12,13,14,15);
}

__device__ __forceinline__ v8f wmma_bf16(v16bf a, v16bf b, v8f c) {
  return __builtin_amdgcn_wmma_f32_16x16x32_bf16(false, a, false, b,
                                                 (short)0, c, false, false);
}

// Two 16x16 tiles (rows of X0 / X1, same n-tile) sharing each B fragment.
__device__ __forceinline__ void wmma_pair(const __bf16* X0, const __bf16* X1,
                                          const __bf16* WtTile, int Kpad,
                                          int lane, v8f& c0, v8f& c1) {
  for (int k = 0; k < Kpad; k += 32) {
    v16bf b = load_b_frag(WtTile, Kpad, k, lane);
    c0 = wmma_bf16(load_a_frag(X0, Kpad, k, lane), b, c0);
    c1 = wmma_bf16(load_a_frag(X1, Kpad, k, lane), b, c1);
  }
}

// ------------------------------------------------------------ prep kernels
__global__ void convert_wt_kernel(const float* __restrict__ W,
                                  __bf16* __restrict__ Wt,
                                  int K, int N, int Kpad) {
  int idx = blockIdx.x * blockDim.x + threadIdx.x;
  if (idx >= N * Kpad) return;
  int n = idx / Kpad, k = idx - n * Kpad;
  Wt[idx] = (__bf16)((k < K) ? W[k * N + n] : 0.0f);
}

__global__ void build_feats_kernel(const float* __restrict__ nf,
                                   const float* __restrict__ emb,
                                   const int* __restrict__ flav,
                                   __bf16* __restrict__ feats, int n) {
  int idx = blockIdx.x * blockDim.x + threadIdx.x;
  if (idx >= n * FDIM) return;
  int node = idx >> 5, c = idx & 31;
  float v = (c < 16) ? nf[node * 16 + c] : emb[flav[node] * 16 + (c - 16)];
  feats[idx] = (__bf16)v;
}

__global__ void zero_f32_kernel(float* __restrict__ p, int n) {
  int idx = blockIdx.x * blockDim.x + threadIdx.x;
  if (idx < n) p[idx] = 0.0f;
}

// ------------------------------------------------------------- edge kernel
// One block = 64 edges, 256 threads (8 waves).
__global__ void edge_mlp_kernel(const __bf16* __restrict__ feats,
                                const __bf16* __restrict__ hid,
                                const float* __restrict__ ef,
                                const int* __restrict__ src,
                                const int* __restrict__ dst,
                                const __bf16* __restrict__ W1t,
                                const float* __restrict__ b1,
                                const __bf16* __restrict__ W2t,
                                const float* __restrict__ b2,
                                float* __restrict__ mail,
                                int n_edges, int K1, int N1) {
  extern __shared__ char smem[];
  __bf16* X = (__bf16*)smem;        // 64 x K1
  __bf16* H = X + 64 * K1;          // 64 x N1
  int* dsts = (int*)(H + 64 * N1);  // 64

  const int tid = threadIdx.x;
  const int e0 = blockIdx.x * 64;
  const bool has_hid = (hid != nullptr);

  // ---- staging: row-per-thread-quad, 16B chunks, no divisions ----
  {
    const int r = tid >> 2;        // 0..63
    const int cq = tid & 3;
    const int e = e0 + r;
    const bool ok = (e < n_edges);
    if (cq == 0) dsts[r] = ok ? dst[e] : -1;
    int d = 0, s = 0;
    if (ok) { d = dst[e]; s = src[e]; }
    const int CH = K1 >> 3;
    for (int c = cq; c < CH; c += 4) {
      int k = c << 3;
      v8bf v = {};
      if (ok) {
        if (has_hid) {             // [f[d](32) h[d](64) f[s](32) h[s](64) ef 0..]
          if (k < 32)       v = *(const v8bf*)(feats + d * FDIM + k);
          else if (k < 96)  v = *(const v8bf*)(hid + d * HDIM + (k - 32));
          else if (k < 128) v = *(const v8bf*)(feats + s * FDIM + (k - 96));
          else if (k < 192) v = *(const v8bf*)(hid + s * HDIM + (k - 128));
          else if (k == 192) v[0] = (__bf16)ef[e];
        } else {                   // [f[d](32) f[s](32) ef 0..]
          if (k < 32)       v = *(const v8bf*)(feats + d * FDIM + k);
          else if (k < 64)  v = *(const v8bf*)(feats + s * FDIM + (k - 32));
          else if (k == 64) v[0] = (__bf16)ef[e];
        }
      }
      *(v8bf*)(X + r * K1 + k) = v;
    }
  }
  __syncthreads();

  const int lane = tid & 31, wave = tid >> 5, nwaves = blockDim.x >> 5;
  const int n = lane & 15, half = lane >> 4;

  // ---- Phase 1: H = relu(X @ W1 + b1) ----
  if (N1 == 128) {                 // 8 n-tiles: 4-way M reuse of B
    for (int nt = wave * 16; nt < 128; nt += nwaves * 16) {
      v8f cc[4] = {{}, {}, {}, {}};
      const __bf16* wt = W1t + nt * K1;
      for (int k = 0; k < K1; k += 32) {
        v16bf b = load_b_frag(wt, K1, k, lane);
#pragma unroll
        for (int t = 0; t < 4; ++t)
          cc[t] = wmma_bf16(load_a_frag(X + t * 16 * K1, K1, k, lane), b, cc[t]);
      }
      float bias = b1[nt + n];
#pragma unroll
      for (int t = 0; t < 4; ++t)
#pragma unroll
        for (int i = 0; i < 8; ++i)
          H[(t * 16 + i + 8 * half) * 128 + nt + n] =
              (__bf16)fmaxf(cc[t][i] + bias, 0.0f);
    }
  } else {                         // N1==64: 4 nt x 2 m-pairs = 8 units
    for (int u = wave; u < 8; u += nwaves) {
      int nt = (u & 3) << 4;
      int mp = (u >> 2) << 5;
      v8f c0 = {}, c1 = {};
      wmma_pair(X + mp * K1, X + (mp + 16) * K1, W1t + nt * K1, K1, lane, c0, c1);
      float bias = b1[nt + n];
#pragma unroll
      for (int i = 0; i < 8; ++i) {
        H[(mp + i + 8 * half) * 64 + nt + n] = (__bf16)fmaxf(c0[i] + bias, 0.0f);
        H[(mp + 16 + i + 8 * half) * 64 + nt + n] = (__bf16)fmaxf(c1[i] + bias, 0.0f);
      }
    }
  }
  __syncthreads();

  // ---- Phase 2: tanh(H @ W2 + b2) -> atomic scatter (4 nt x 2 m-pairs) ----
  for (int u = wave; u < 8; u += nwaves) {
    int nt = (u & 3) << 4;
    int mp = (u >> 2) << 5;
    v8f c0 = {}, c1 = {};
    wmma_pair(H + mp * N1, H + (mp + 16) * N1, W2t + nt * N1, N1, lane, c0, c1);
    float bias = b2[nt + n];
#pragma unroll
    for (int i = 0; i < 8; ++i) {
      int d0 = dsts[mp + i + 8 * half];
      if (d0 >= 0) atomicAdd(&mail[d0 * HDIM + nt + n], tanhf(c0[i] + bias));
      int d1 = dsts[mp + 16 + i + 8 * half];
      if (d1 >= 0) atomicAdd(&mail[d1 * HDIM + nt + n], tanhf(c1[i] + bias));
    }
  }
}

// ------------------------------------------------------------- node kernel
// One block = 64 nodes. r1 = mlp(node_in), r2 = mlp(mail), hid = norm([r1 r2]).
__global__ void node_mlp_kernel(const __bf16* __restrict__ feats,
                                const __bf16* __restrict__ hid_in,
                                const float* __restrict__ mail,
                                const __bf16* n1W1t, const float* n1b1,
                                const __bf16* n1W2t, const float* n1b2,
                                const __bf16* n2W1t, const float* n2b1,
                                const __bf16* n2W2t, const float* n2b2,
                                __bf16* __restrict__ hid_out,
                                int n_nodes, int Kn, int M1) {
  extern __shared__ char smem[];
  __bf16* X1 = (__bf16*)smem;            // 64 x Kn
  __bf16* X2 = X1 + 64 * Kn;             // 64 x 64
  __bf16* H1 = X2 + 64 * 64;             // 64 x M1
  __bf16* H2 = H1 + 64 * M1;             // 64 x 32
  float*  R  = (float*)(H2 + 64 * 32);   // 64 x 64

  const int tid = threadIdx.x;
  const int v0 = blockIdx.x * 64;

  {  // stage X1 (node_in) and X2 (mail), row-per-thread-quad
    const int r = tid >> 2;
    const int cq = tid & 3;
    const int node = v0 + r;
    const bool ok = (node < n_nodes);
    const int CH = Kn >> 3;
    for (int c = cq; c < CH; c += 4) {
      int k = c << 3;
      v8bf v = {};
      if (ok)
        v = (k < 32) ? *(const v8bf*)(feats + node * FDIM + k)
                     : *(const v8bf*)(hid_in + node * HDIM + (k - 32));
      *(v8bf*)(X1 + r * Kn + k) = v;
    }
    const int kb = cq << 4;
    for (int j = 0; j < 16; ++j)
      X2[r * 64 + kb + j] = (__bf16)(ok ? mail[node * HDIM + kb + j] : 0.0f);
  }
  __syncthreads();

  const int lane = tid & 31, wave = tid >> 5, nwaves = blockDim.x >> 5;
  const int n = lane & 15, half = lane >> 4;

  // ---- Phase A: H1 = relu(X1@n1W1+b), H2 = relu(X2@n2W1+b) fused ----
  {
    const int ntiles1 = M1 >> 4;               // 2 or 4
    const int unitsH1 = ntiles1 * 2;           // 4 or 8
    for (int u = wave; u < unitsH1 + 4; u += nwaves) {
      v8f c0 = {}, c1 = {};
      if (u < unitsH1) {
        int nt = (u % ntiles1) << 4;
        int mp = (u / ntiles1) << 5;
        wmma_pair(X1 + mp * Kn, X1 + (mp + 16) * Kn, n1W1t + nt * Kn, Kn,
                  lane, c0, c1);
        float bias = n1b1[nt + n];
#pragma unroll
        for (int i = 0; i < 8; ++i) {
          H1[(mp + i + 8 * half) * M1 + nt + n] = (__bf16)fmaxf(c0[i] + bias, 0.0f);
          H1[(mp + 16 + i + 8 * half) * M1 + nt + n] = (__bf16)fmaxf(c1[i] + bias, 0.0f);
        }
      } else {
        int v = u - unitsH1;                   // 0..3
        int nt = (v & 1) << 4;
        int mp = (v >> 1) << 5;
        wmma_pair(X2 + mp * 64, X2 + (mp + 16) * 64, n2W1t + nt * 64, 64,
                  lane, c0, c1);
        float bias = n2b1[nt + n];
#pragma unroll
        for (int i = 0; i < 8; ++i) {
          H2[(mp + i + 8 * half) * 32 + nt + n] = (__bf16)fmaxf(c0[i] + bias, 0.0f);
          H2[(mp + 16 + i + 8 * half) * 32 + nt + n] = (__bf16)fmaxf(c1[i] + bias, 0.0f);
        }
      }
    }
  }
  __syncthreads();

  // ---- Phase B: R[:,0:32] = tanh(H1@n1W2+b); R[:,32:64] = tanh(H2@n2W2+b) ----
  for (int u = wave; u < 8; u += nwaves) {
    int v = u & 3;
    int nt = (v & 1) << 4;
    int mp = (v >> 1) << 5;
    v8f c0 = {}, c1 = {};
    float bias;
    int col;
    if (u < 4) {
      wmma_pair(H1 + mp * M1, H1 + (mp + 16) * M1, n1W2t + nt * M1, M1,
                lane, c0, c1);
      bias = n1b2[nt + n]; col = nt;
    } else {
      wmma_pair(H2 + mp * 32, H2 + (mp + 16) * 32, n2W2t + nt * 32, 32,
                lane, c0, c1);
      bias = n2b2[nt + n]; col = 32 + nt;
    }
#pragma unroll
    for (int i = 0; i < 8; ++i) {
      R[(mp + i + 8 * half) * 64 + col + n] = tanhf(c0[i] + bias);
      R[(mp + 16 + i + 8 * half) * 64 + col + n] = tanhf(c1[i] + bias);
    }
  }
  __syncthreads();

  if (tid < 64) {  // row L2-normalize
    int node = v0 + tid;
    if (node < n_nodes) {
      float s = 0.0f;
#pragma unroll
      for (int k = 0; k < 64; ++k) { float x = R[tid * 64 + k]; s += x * x; }
      float inv = 1.0f / sqrtf(s);
#pragma unroll
      for (int k = 0; k < 64; ++k)
        hid_out[node * HDIM + k] = (__bf16)(R[tid * 64 + k] * inv);
    }
  }
}

// -------------------------------------------------------------- jet kernel
// One block = 64 nodes. h=[feats,hid](96) -> relu(128) -> relu(64) -> scalar
__global__ void jet_kernel(const __bf16* __restrict__ feats,
                           const __bf16* __restrict__ hid,
                           const __bf16* W1t, const float* b1,
                           const __bf16* W2t, const float* b2,
                           const float* W3, const float* b3,
                           float* __restrict__ out, int n_nodes) {
  extern __shared__ char smem[];
  __bf16* X  = (__bf16*)smem;              // 64 x 96
  __bf16* H1 = X + 64 * 96;                // 64 x 128
  float*  H2 = (float*)(H1 + 64 * 128);    // 64 x 64

  const int tid = threadIdx.x;
  const int v0 = blockIdx.x * 64;

  {
    const int r = tid >> 2;
    const int cq = tid & 3;
    const int node = v0 + r;
    const bool ok = (node < n_nodes);
    for (int c = cq; c < 12; c += 4) {
      int k = c << 3;
      v8bf v = {};
      if (ok)
        v = (k < 32) ? *(const v8bf*)(feats + node * FDIM + k)
                     : *(const v8bf*)(hid + node * HDIM + (k - 32));
      *(v8bf*)(X + r * 96 + k) = v;
    }
  }
  __syncthreads();

  const int lane = tid & 31, wave = tid >> 5, nwaves = blockDim.x >> 5;
  const int n = lane & 15, half = lane >> 4;

  // H1 = relu(X @ W1 + b1): 8 n-tiles, 4-way M reuse of B.
  for (int nt = wave * 16; nt < 128; nt += nwaves * 16) {
    v8f cc[4] = {{}, {}, {}, {}};
    const __bf16* wt = W1t + nt * 96;
    for (int k = 0; k < 96; k += 32) {
      v16bf b = load_b_frag(wt, 96, k, lane);
#pragma unroll
      for (int t = 0; t < 4; ++t)
        cc[t] = wmma_bf16(load_a_frag(X + t * 16 * 96, 96, k, lane), b, cc[t]);
    }
    float bias = b1[nt + n];
#pragma unroll
    for (int t = 0; t < 4; ++t)
#pragma unroll
      for (int i = 0; i < 8; ++i)
        H1[(t * 16 + i + 8 * half) * 128 + nt + n] =
            (__bf16)fmaxf(cc[t][i] + bias, 0.0f);
  }
  __syncthreads();

  // H2 = relu(H1 @ W2 + b2): 4 nt x 2 m-pairs = 8 units
  for (int u = wave; u < 8; u += nwaves) {
    int nt = (u & 3) << 4;
    int mp = (u >> 2) << 5;
    v8f c0 = {}, c1 = {};
    wmma_pair(H1 + mp * 128, H1 + (mp + 16) * 128, W2t + nt * 128, 128,
              lane, c0, c1);
    float bias = b2[nt + n];
#pragma unroll
    for (int i = 0; i < 8; ++i) {
      H2[(mp + i + 8 * half) * 64 + nt + n] = fmaxf(c0[i] + bias, 0.0f);
      H2[(mp + 16 + i + 8 * half) * 64 + nt + n] = fmaxf(c1[i] + bias, 0.0f);
    }
  }
  __syncthreads();

  if (tid < 64) {
    int node = v0 + tid;
    if (node < n_nodes) {
      float s = b3[0];
#pragma unroll
      for (int k = 0; k < 64; ++k) s += H2[tid * 64 + k] * W3[k];
      out[node] = s;
    }
  }
}

// ---------------------------------------------------------------- launcher
extern "C" void kernel_launch(void* const* d_in, const int* in_sizes, int n_in,
                              void* d_out, int out_size, void* d_ws, size_t ws_size,
                              hipStream_t stream) {
  (void)in_sizes; (void)n_in; (void)out_size; (void)ws_size;

  const float* node_feats = (const float*)d_in[0];
  const float* edge_feat  = (const float*)d_in[1];
  const float* emb        = (const float*)d_in[2];
  const float* L[3][12];
  for (int l = 0; l < 3; ++l)
    for (int j = 0; j < 12; ++j) L[l][j] = (const float*)d_in[3 + 12 * l + j];
  const float* jW1 = (const float*)d_in[39];
  const float* jb1 = (const float*)d_in[40];
  const float* jW2 = (const float*)d_in[41];
  const float* jb2 = (const float*)d_in[42];
  const float* jW3 = (const float*)d_in[43];
  const float* jb3 = (const float*)d_in[44];
  const int* flav = (const int*)d_in[45];
  const int* srcI = (const int*)d_in[46];
  const int* dstI = (const int*)d_in[47];
  float* out = (float*)d_out;

  char* ws = (char*)d_ws;
  size_t off = 0;
  auto alloc = [&](size_t bytes) -> char* {
    char* p = ws + off;
    off += (bytes + 255) & ~(size_t)255;
    return p;
  };
  __bf16* featsB = (__bf16*)alloc((size_t)NN * FDIM * 2);
  __bf16* hidA   = (__bf16*)alloc((size_t)NN * HDIM * 2);
  __bf16* hidB   = (__bf16*)alloc((size_t)NN * HDIM * 2);
  float*  mail   = (float*)alloc((size_t)NN * HDIM * 4);

  const int e_in[3]  = {65, 193, 193};
  const int e_k[3]   = {96, 224, 224};   // K padded to 32
  const int e_mid[3] = {64, 128, 128};
  const int n_in_[3] = {32, 96, 96};
  const int n_mid[3] = {32, 64, 64};

  __bf16 *eW1t[3], *eW2t[3], *n1W1t[3], *n1W2t[3], *n2W1t[3], *n2W2t[3];
  for (int l = 0; l < 3; ++l) {
    eW1t[l]  = (__bf16*)alloc((size_t)e_mid[l] * e_k[l] * 2);
    eW2t[l]  = (__bf16*)alloc((size_t)HDIM * e_mid[l] * 2);
    n1W1t[l] = (__bf16*)alloc((size_t)n_mid[l] * n_in_[l] * 2);
    n1W2t[l] = (__bf16*)alloc((size_t)32 * n_mid[l] * 2);
    n2W1t[l] = (__bf16*)alloc((size_t)32 * 64 * 2);
    n2W2t[l] = (__bf16*)alloc((size_t)32 * 32 * 2);
  }
  __bf16* jW1t = (__bf16*)alloc((size_t)128 * 96 * 2);
  __bf16* jW2t = (__bf16*)alloc((size_t)64 * 128 * 2);

  auto conv = [&](const float* W, __bf16* Wt, int K, int N, int Kpad) {
    int total = N * Kpad;
    convert_wt_kernel<<<(total + 255) / 256, 256, 0, stream>>>(W, Wt, K, N, Kpad);
  };
  for (int l = 0; l < 3; ++l) {
    conv(L[l][0],  eW1t[l],  e_in[l],  e_mid[l], e_k[l]);
    conv(L[l][2],  eW2t[l],  e_mid[l], HDIM,     e_mid[l]);
    conv(L[l][4],  n1W1t[l], n_in_[l], n_mid[l], n_in_[l]);
    conv(L[l][6],  n1W2t[l], n_mid[l], 32,       n_mid[l]);
    conv(L[l][8],  n2W1t[l], 64,       32,       64);
    conv(L[l][10], n2W2t[l], 32,       32,       32);
  }
  conv(jW1, jW1t, 96, 128, 96);
  conv(jW2, jW2t, 128, 64, 128);

  build_feats_kernel<<<(NN * FDIM + 255) / 256, 256, 0, stream>>>(
      node_feats, emb, flav, featsB, NN);

  const __bf16* hid_in = nullptr;
  __bf16* hid_out = hidA;
  for (int l = 0; l < 3; ++l) {
    zero_f32_kernel<<<(NN * HDIM + 255) / 256, 256, 0, stream>>>(mail, NN * HDIM);

    size_t smemE = (size_t)64 * (e_k[l] + e_mid[l]) * 2 + 64 * 4;
    edge_mlp_kernel<<<(NE + 63) / 64, 256, smemE, stream>>>(
        featsB, hid_in, edge_feat, srcI, dstI,
        eW1t[l], L[l][1], eW2t[l], L[l][3], mail, NE, e_k[l], e_mid[l]);

    size_t smemN = (size_t)64 * (n_in_[l] + 64 + n_mid[l] + 32) * 2
                 + (size_t)64 * 64 * 4;
    node_mlp_kernel<<<(NN + 63) / 64, 256, smemN, stream>>>(
        featsB, hid_in, mail,
        n1W1t[l], L[l][5], n1W2t[l], L[l][7],
        n2W1t[l], L[l][9], n2W2t[l], L[l][11],
        hid_out, NN, n_in_[l], n_mid[l]);

    hid_in = hid_out;
    hid_out = (hid_out == hidA) ? hidB : hidA;
  }

  size_t smemJ = (size_t)64 * 96 * 2 + (size_t)64 * 128 * 2 + (size_t)64 * 64 * 4;
  jet_kernel<<<(NN + 63) / 64, 256, smemJ, stream>>>(
      featsB, hid_in, jW1t, jb1, jW2t, jb2, jW3, jb3, out, NN);
}
